// Quantize_1692217114653
// MI455X (gfx1250) — matrix operand order repaired and checked by reading, not compiled, not run on hardware
//
#include <hip/hip_runtime.h>

#define NROWS   131072      // 32*64*64
#define DIM     64
#define NC      512
#define DECAYF  0.99f
#define OMDEC   0.01f
#define EPSV    1e-5f

// output float offsets (tuple concatenated flat)
#define O_QUANT 0           // 8388608
#define O_DIFF  8388608     // 1
#define O_IND   8388609     // 131072
#define O_NCS   8519681     // 512
#define O_NEA   8520193     // 32768
#define O_NE    8552961     // 32768

// workspace float offsets
#define WS_DIFF   0
#define WS_NSUM   1
#define WS_ESCORE 16        // 512
#define WS_COUNT  528       // 512
#define WS_EMBSUM 1040      // 32768
#define WS_EMBBF  33808     // 32768 ushorts (bf16 embed, transposed [j][d])
#define WS_ZERO_N 33808

typedef __attribute__((ext_vector_type(16))) __bf16 v16bf;
typedef __attribute__((ext_vector_type(8)))  float  v8f;

union BFVec { v16bf v; unsigned short u[16]; uint4 q[2]; };

__device__ __forceinline__ unsigned short f2bf(float f) {
  unsigned int u = __float_as_uint(f);
  u += 0x7FFFu + ((u >> 16) & 1u);           // round-to-nearest-even
  return (unsigned short)(u >> 16);
}

__global__ void k_zero(float* ws, int n) {
  int i = blockIdx.x * blockDim.x + threadIdx.x;
  if (i < n) ws[i] = 0.0f;
}

// ||E_j||^2 in exact f32 (bias term for the argmin)
__global__ void k_prep_escore(const float* __restrict__ embed, float* __restrict__ ws) {
  int j = blockIdx.x * blockDim.x + threadIdx.x;
  if (j >= NC) return;
  float s = 0.0f;
  for (int d = 0; d < DIM; ++d) { float e = embed[d * NC + j]; s = fmaf(e, e, s); }
  ws[WS_ESCORE + j] = s;
}

// transposed bf16 codebook: embT[j*64 + d] so B fragments are contiguous 16B loads
__global__ void k_prep_bf(const float* __restrict__ embed, float* __restrict__ ws) {
  int i = blockIdx.x * blockDim.x + threadIdx.x;
  if (i >= NC * DIM) return;
  int j = i >> 6, d = i & 63;
  ((unsigned short*)(ws + WS_EMBBF))[i] = f2bf(embed[d * NC + j]);
}

__global__ __launch_bounds__(256) void k_vq(const float* __restrict__ in,
                                            const float* __restrict__ embed,
                                            float* __restrict__ ws,
                                            float* __restrict__ out) {
  const int lane = threadIdx.x & 31;
  const int wave = threadIdx.x >> 5;
  const int m0   = blockIdx.x * 128 + wave * 16;   // 16-row tile per wave

  const unsigned short* embT = (const unsigned short*)(ws + WS_EMBBF);
  const float* esc   = ws + WS_ESCORE;
  float* wcount      = ws + WS_COUNT;
  float* wsum        = ws + WS_EMBSUM;

  // ---- A fragments: 16x32 bf16 per K-tile, ISA 16-bit A layout ----
  // Per lane the 16 elements are two contiguous 8-float runs:
  //   e=0..7  -> k = khalf + 32t + e
  //   e=8..15 -> k = khalf + 32t + 16 + (e-8)
  const int arow  = m0 + (lane & 15);
  const int khalf = (lane >> 4) << 3;              // 0 or 8
  const float* arp = in + arow * DIM + khalf;
  BFVec A[2];
  #pragma unroll
  for (int t = 0; t < 2; ++t) {
    float4 r0 = *(const float4*)(arp + 32 * t);
    float4 r1 = *(const float4*)(arp + 32 * t + 4);
    float4 r2 = *(const float4*)(arp + 32 * t + 16);
    float4 r3 = *(const float4*)(arp + 32 * t + 20);
    A[t].u[0]  = f2bf(r0.x); A[t].u[1]  = f2bf(r0.y);
    A[t].u[2]  = f2bf(r0.z); A[t].u[3]  = f2bf(r0.w);
    A[t].u[4]  = f2bf(r1.x); A[t].u[5]  = f2bf(r1.y);
    A[t].u[6]  = f2bf(r1.z); A[t].u[7]  = f2bf(r1.w);
    A[t].u[8]  = f2bf(r2.x); A[t].u[9]  = f2bf(r2.y);
    A[t].u[10] = f2bf(r2.z); A[t].u[11] = f2bf(r2.w);
    A[t].u[12] = f2bf(r3.x); A[t].u[13] = f2bf(r3.y);
    A[t].u[14] = f2bf(r3.z); A[t].u[15] = f2bf(r3.w);
  }

  float minv[8]; int minc[8];
  #pragma unroll
  for (int r = 0; r < 8; ++r) { minv[r] = 3.4e38f; minc[r] = 0; }

  const int colb = lane & 15;
  const int kofs = (lane >> 4) << 3;               // 0 or 8 (B layout K half)

  // 2 N-tiles per iteration: two independent WMMA accumulator chains so the
  // compiler fills the WMMA->VALU hazard slots instead of emitting v_nops.
  for (int nt = 0; nt < 32; nt += 2) {
    int colA = nt * 16 + colb;
    int colB = colA + 16;
    const uint4* pa = (const uint4*)(embT + colA * DIM + kofs);
    const uint4* pb = (const uint4*)(embT + colB * DIM + kofs);
    BFVec Ba0, Ba1, Bb0, Bb1;
    Ba0.q[0] = pa[0]; Ba0.q[1] = pa[2];
    Ba1.q[0] = pa[4]; Ba1.q[1] = pa[6];   // +32 ushorts == +4 uint4
    Bb0.q[0] = pb[0]; Bb0.q[1] = pb[2];
    Bb1.q[0] = pb[4]; Bb1.q[1] = pb[6];
    float esA = esc[colA];
    float esB = esc[colB];

    v8f acc0 = {0.f, 0.f, 0.f, 0.f, 0.f, 0.f, 0.f, 0.f};
    v8f acc1 = {0.f, 0.f, 0.f, 0.f, 0.f, 0.f, 0.f, 0.f};
    acc0 = __builtin_amdgcn_wmma_f32_16x16x32_bf16(false, A[0].v, false, Ba0.v,
                                                   (short)0, acc0, false, false);
    acc1 = __builtin_amdgcn_wmma_f32_16x16x32_bf16(false, A[0].v, false, Bb0.v,
                                                   (short)0, acc1, false, false);
    acc0 = __builtin_amdgcn_wmma_f32_16x16x32_bf16(false, A[1].v, false, Ba1.v,
                                                   (short)0, acc0, false, false);
    acc1 = __builtin_amdgcn_wmma_f32_16x16x32_bf16(false, A[1].v, false, Bb1.v,
                                                   (short)0, acc1, false, false);

    #pragma unroll
    for (int r = 0; r < 8; ++r) {
      float s0 = fmaf(-2.0f, acc0[r], esA);        // ||E||^2 - 2 f.E  (||f||^2 dropped)
      if (s0 < minv[r]) { minv[r] = s0; minc[r] = colA; }
      float s1 = fmaf(-2.0f, acc1[r], esB);
      if (s1 < minv[r]) { minv[r] = s1; minc[r] = colB; }
    }
  }

  // ---- argmin reduce across the 16-lane C-layout groups ----
  #pragma unroll
  for (int m = 1; m < 16; m <<= 1) {
    #pragma unroll
    for (int r = 0; r < 8; ++r) {
      float ov = __shfl_xor(minv[r], m, 32);
      int   oc = __shfl_xor(minc[r], m, 32);
      if (ov < minv[r] || (ov == minv[r] && oc < minc[r])) { minv[r] = ov; minc[r] = oc; }
    }
  }

  // lanes 0..7 own rows m0+0..7, lanes 16..23 own rows m0+8..15
  if ((lane & 15) < 8) {
    int r = lane & 15;
    int row = m0 + r + ((lane >> 4) << 3);
    int j = minc[r];
    out[O_IND + row] = (float)j;
    unsafeAtomicAdd(&wcount[j], 1.0f);
  }

  // ---- quantize lookup, diff partials, embed_sum scatter ----
  float dacc = 0.0f;
  #pragma unroll
  for (int rr = 0; rr < 16; ++rr) {
    int j   = __shfl(minc[rr & 7], (rr < 8) ? 0 : 16, 32);
    int row = m0 + rr;
    #pragma unroll
    for (int h = 0; h < 2; ++h) {
      int d   = lane + 32 * h;
      float e = embed[d * NC + j];
      float x = in[row * DIM + d];
      out[O_QUANT + row * DIM + d] = e;            // quantize_st == quantize numerically
      float df = e - x;
      dacc = fmaf(df, df, dacc);
      unsafeAtomicAdd(&wsum[d * NC + j], x);
    }
  }
  #pragma unroll
  for (int m = 1; m < 32; m <<= 1) dacc += __shfl_xor(dacc, m, 32);
  if (lane == 0) unsafeAtomicAdd(ws + WS_DIFF, dacc);
}

__global__ void k_fin_cs(const float* __restrict__ cs_in, float* __restrict__ ws,
                         float* __restrict__ out) {
  __shared__ float red[NC];
  int j = threadIdx.x;
  float c = cs_in[j] * DECAYF + OMDEC * ws[WS_COUNT + j];
  out[O_NCS + j] = c;
  red[j] = c;
  __syncthreads();
  for (int s = NC / 2; s > 0; s >>= 1) {
    if (j < s) red[j] += red[j + s];
    __syncthreads();
  }
  if (j == 0) {
    ws[WS_NSUM] = red[0];
    out[O_DIFF] = ws[WS_DIFF] * (1.0f / 8388608.0f);
  }
}

__global__ void k_fin_embed(const float* __restrict__ eavg, const float* __restrict__ ws,
                            float* __restrict__ out) {
  int i = blockIdx.x * blockDim.x + threadIdx.x;
  if (i >= DIM * NC) return;
  int j = i & (NC - 1);
  float nea = eavg[i] * DECAYF + OMDEC * ws[WS_EMBSUM + i];
  out[O_NEA + i] = nea;
  float n   = ws[WS_NSUM];
  float ncs = out[O_NCS + j];
  float cs  = (ncs + EPSV) / (n + (float)NC * EPSV) * n;
  out[O_NE + i] = nea / cs;
}

extern "C" void kernel_launch(void* const* d_in, const int* in_sizes, int n_in,
                              void* d_out, int out_size, void* d_ws, size_t ws_size,
                              hipStream_t stream) {
  const float* in    = (const float*)d_in[0];   // [32,64,64,64]
  const float* embed = (const float*)d_in[1];   // [64,512]
  const float* csz   = (const float*)d_in[2];   // [512]
  const float* eavg  = (const float*)d_in[3];   // [64,512]
  float* out = (float*)d_out;
  float* ws  = (float*)d_ws;

  k_zero<<<(WS_ZERO_N + 255) / 256, 256, 0, stream>>>(ws, WS_ZERO_N);
  k_prep_escore<<<(NC + 255) / 256, 256, 0, stream>>>(embed, ws);
  k_prep_bf<<<(NC * DIM + 255) / 256, 256, 0, stream>>>(embed, ws);
  k_vq<<<NROWS / 128, 256, 0, stream>>>(in, embed, ws, out);
  k_fin_cs<<<1, NC, 0, stream>>>(csz, ws, out);
  k_fin_embed<<<(DIM * NC + 255) / 256, 256, 0, stream>>>(eavg, ws, out);
}
